// FNOGNO_26517128085829
// MI455X (gfx1250) — matrix operand order repaired and checked by reading, not compiled, not run on hardware
//
#include <hip/hip_runtime.h>
#include <hip/hip_bf16.h>
#include <math.h>

// ---------------- problem constants ----------------
#define GSZ   32
#define NG    (GSZ*GSZ*GSZ)      // 32768 latent grid points
#define CCH   86                 // FNO channels
#define CP    96                 // padded channel stride (mult of 32)
#define MMODE 8                  // kept Fourier modes per dim
#define NL    4                  // FNO layers
#define KNBR  32                 // neighbor cap
#define NIN   4096               // query points
#define EMB3  48                 // 3 coords * 16 embed dims
#define RADF  0.055f
#define CHQ   512                // queries per GNO chunk
#define NCHUNK (NIN/CHQ)         // 8
#define ROWSC (CHQ*KNBR)         // 16384 edge rows per chunk
#define PI_F  3.14159265358979f

// GEMM tiling: block 256x32, 8 waves stacked in M, each wave owns 32x32
#define BM 256
#define BN 32
#define BK 32

typedef _Float16 h16;
typedef __attribute__((ext_vector_type(16))) _Float16 v16h;
typedef __attribute__((ext_vector_type(8)))  _Float16 v8h;
typedef __attribute__((ext_vector_type(8)))  float    v8f;
typedef __attribute__((ext_vector_type(4)))  int      v4i;

// CDNA5 async global->LDS copy path (guarded; falls back to sync copy)
#if defined(__has_builtin)
#if __has_builtin(__builtin_amdgcn_global_load_async_to_lds_b128) && \
    __has_builtin(__builtin_amdgcn_s_wait_asynccnt)
#define USE_ASYNC_LDS 1
#endif
#endif

__device__ __forceinline__ float gelu_exact(float x) {
  return 0.5f * x * (1.0f + erff(x * 0.70710678118654752f));
}

// two contiguous 16B LDS chunks -> one 16-element f16 fragment
__device__ __forceinline__ v16h frag2(const h16* p0, const h16* p1) {
  v8h lo = *(const v8h*)p0;
  v8h hi = *(const v8h*)p1;
  return __builtin_shufflevector(lo, hi, 0,1,2,3,4,5,6,7,8,9,10,11,12,13,14,15);
}

__device__ __forceinline__ void store_tile(const v8f& acc, h16* __restrict__ D,
                                           const h16* __restrict__ bias,
                                           const h16* __restrict__ addend,
                                           int act, int rowb, int colb, int N, int lane)
{
  const int col = colb + (lane & 15);
  const int rb  = rowb + ((lane < 16) ? 0 : 8);
  const float bv = (float)bias[col];
#pragma unroll
  for (int r = 0; r < 8; ++r) {
    int row = rb + r;
    float v = acc[r] + bv;
    if (addend) v += (float)addend[(size_t)row * N + col];
    if (act)    v = gelu_exact(v);
    D[(size_t)row * N + col] = (h16)v;
  }
}

// =====================================================================
// WMMA GEMM (all-f16 operands, f32 accumulation):
//   D[M,N] = act( A[M,K] * B[K,N] + bias + addend )
// REQUIRES: M % 256 == 0, N % 32 == 0, K % 32 == 0, 16B alignment.
// A tile staged as pure 16B copies (async-to-LDS when available, tracked
// by ASYNCcnt + s_wait_asynccnt); B tile staged transposed; fragments
// are ds_load_b128 pairs; 4 back-to-back v_wmma per K-slab per wave.
// =====================================================================
__global__ __launch_bounds__(256)
void gemm_f16_wmma(const h16* __restrict__ A, const h16* __restrict__ B,
                   const h16* __restrict__ bias, const h16* __restrict__ addend,
                   h16* __restrict__ D, int M, int N, int K, int act)
{
  (void)M;
  __shared__ __align__(16) h16 lA [BM][BK];   // 16 KB, row-major
  __shared__ __align__(16) h16 lBt[BN][BK];   // 2 KB, transposed B

  const int wave = threadIdx.x >> 5;
  const int lane = threadIdx.x & 31;
  const int wm = wave << 5;              // wave row offset 0..224
  const int m0 = blockIdx.x * BM;
  const int n0 = blockIdx.y * BN;

  v8f acc00 = {0.f,0.f,0.f,0.f,0.f,0.f,0.f,0.f};
  v8f acc01 = acc00, acc10 = acc00, acc11 = acc00;

  for (int k0 = 0; k0 < K; k0 += BK) {
    // speculative prefetch of next A slab (global_prefetch_b8 path)
    if (k0 + BK < K) {
      __builtin_prefetch(A + (size_t)(m0 + (int)(threadIdx.x >> 1)) * K + (k0 + BK), 0, 1);
    }
    // ---- stage A tile 256x32 f16: 1024 x 16B chunks, 4/thread ----
#pragma unroll
    for (int p = 0; p < 4; ++p) {
      int slot = (int)threadIdx.x + p * 256;   // 0..1023
      int r  = slot >> 2;                      // 0..255
      int cg = (slot & 3) << 3;                // 0,8,16,24
      const h16* g = A + (size_t)(m0 + r) * K + (k0 + cg);
#ifdef USE_ASYNC_LDS
      __builtin_amdgcn_global_load_async_to_lds_b128(
          (__attribute__((address_space(1))) v4i*)g,
          (__attribute__((address_space(3))) v4i*)&lA[r][cg], 0, 0);
#else
      *(v8h*)&lA[r][cg] = *(const v8h*)g;
#endif
    }
    // ---- stage B tile 32x32 transposed: threads 0..127, strided u16 ----
    if (threadIdx.x < 128) {
      int n  = (int)threadIdx.x & 31;
      int kg = ((int)threadIdx.x >> 5) << 3;   // 0,8,16,24
      const h16* bp = B + (size_t)(k0 + kg) * N + (n0 + n);
      v8h h;
#pragma unroll
      for (int j = 0; j < 8; ++j) h[j] = bp[(size_t)j * N];
      *(v8h*)&lBt[n][kg] = h;
    }
#ifdef USE_ASYNC_LDS
    __builtin_amdgcn_s_wait_asynccnt(0);
#endif
    __syncthreads();

    // ---- fragments (ds_load_b128 pairs) ----
    const int kbA = (lane < 16) ? 0 : 8;
    const int ra  = wm + (lane & 15);
    v16h a0 = frag2(&lA[ra     ][kbA], &lA[ra     ][kbA + 16]);
    v16h a1 = frag2(&lA[ra + 16][kbA], &lA[ra + 16][kbA + 16]);
    const int kbB = (lane < 16) ? 0 : 8;
    const int cb  = lane & 15;
    v16h b0 = frag2(&lBt[cb     ][kbB], &lBt[cb     ][kbB + 16]);
    v16h b1 = frag2(&lBt[cb + 16][kbB], &lBt[cb + 16][kbB + 16]);

    acc00 = __builtin_amdgcn_wmma_f32_16x16x32_f16(false, a0, false, b0, (short)0, acc00, false, false);
    acc01 = __builtin_amdgcn_wmma_f32_16x16x32_f16(false, a0, false, b1, (short)0, acc01, false, false);
    acc10 = __builtin_amdgcn_wmma_f32_16x16x32_f16(false, a1, false, b0, (short)0, acc10, false, false);
    acc11 = __builtin_amdgcn_wmma_f32_16x16x32_f16(false, a1, false, b1, (short)0, acc11, false, false);
    __syncthreads();
  }

  store_tile(acc00, D, bias, addend, act, m0 + wm,      n0,      N, lane);
  store_tile(acc01, D, bias, addend, act, m0 + wm,      n0 + 16, N, lane);
  store_tile(acc10, D, bias, addend, act, m0 + wm + 16, n0,      N, lane);
  store_tile(acc11, D, bias, addend, act, m0 + wm + 16, n0 + 16, N, lane);
}

// =====================================================================
// Padding / staging kernels
// =====================================================================

// batched zero-pad + f32->f16: src [B, rows, cols] -> dst [B, prows, pcols]
__global__ void k_padbatch(const float* __restrict__ src, h16* __restrict__ dst,
                           int nb, int rows, int cols, int prows, int pcols)
{
  int n = blockIdx.x * blockDim.x + threadIdx.x;
  int per = prows * pcols;
  if (n >= nb * per) return;
  int b = n / per, rem = n % per;
  int r = rem / pcols, c = rem % pcols;
  dst[n] = (h16)((r < rows && c < cols) ? src[((size_t)b * rows + r) * cols + c] : 0.0f);
}

// =====================================================================
// Elementwise / DFT / graph kernels
// =====================================================================

// PositionalEmbedding: per point, [cos(x*f0..7), sin(x*f0..7), y..., z...]
__global__ void k_embed(const float* __restrict__ pts, h16* __restrict__ out, int npts)
{
  int n = blockIdx.x * blockDim.x + threadIdx.x;
  if (n >= npts * EMB3) return;
  int p = n / EMB3, e = n % EMB3;
  int cidx = e >> 4;          // coordinate 0..2
  int w = e & 15;             // 0..15 within the 16-dim embedding
  int k = (w < 8) ? w : (w - 8);
  float freq = __powf(1.0e-4f, (float)k * 0.125f);
  float ang = pts[p * 3 + cidx] * freq;
  out[n] = (h16)((w < 8) ? __cosf(ang) : __sinf(ang));
}

// FNO input channels padded to 32: [df0, df1, x, y, z, 0...]
__global__ void k_build_vin(const float* __restrict__ df, const float* __restrict__ xout,
                            h16* __restrict__ vin)
{
  int n = blockIdx.x * blockDim.x + threadIdx.x;
  if (n >= NG * 32) return;
  int p = n >> 5, ch = n & 31;
  float v = 0.0f;
  if (ch < 2)      v = df[ch * NG + p];
  else if (ch < 5) v = xout[p * 3 + (ch - 2)];
  vin[n] = (h16)v;
}

// --- truncated forward DFT, separable (only kept modes computed) ---
// v[x][y][z][c](f16, stride CP) -> T1[x][y][kz(0..7)][c]  (f32 complex)
__global__ void k_dft_z(const h16* __restrict__ v, float* __restrict__ T1)
{
  int n = blockIdx.x * blockDim.x + threadIdx.x;
  if (n >= GSZ * GSZ * MMODE * CCH) return;
  int c = n % CCH; int t = n / CCH;
  int kz = t & 7; t >>= 3;
  int y = t & 31; int x = t >> 5;
  float sre = 0.f, sim = 0.f;
  float step = -(PI_F / 16.0f) * (float)kz;
  for (int z = 0; z < GSZ; ++z) {
    float sn, cs; __sincosf(step * (float)z, &sn, &cs);
    float vv = (float)v[((size_t)((x * GSZ + y) * GSZ + z)) * CP + c];
    sre += vv * cs; sim += vv * sn;
  }
  T1[2 * n] = sre; T1[2 * n + 1] = sim;
}

// T1[x][y][kz][c] -> T2[x][ky16][kz][c]; ky16<8 => +ky, else negative modes (ky+16)
__global__ void k_dft_y(const float* __restrict__ T1, float* __restrict__ T2)
{
  int n = blockIdx.x * blockDim.x + threadIdx.x;
  if (n >= GSZ * 16 * MMODE * CCH) return;
  int c = n % CCH; int t = n / CCH;
  int kz = t & 7; t >>= 3;
  int ky = t & 15; int x = t >> 4;
  int fy = (ky < 8) ? ky : (ky + 16);
  float are = 0.f, aim = 0.f;
  for (int y = 0; y < GSZ; ++y) {
    float sn, cs; __sincosf(-(PI_F / 16.0f) * (float)(fy * y), &sn, &cs);
    size_t s = 2 * (((size_t)(x * GSZ + y) * MMODE + kz) * CCH + c);
    float tr = T1[s], ti = T1[s + 1];
    are += tr * cs - ti * sn;
    aim += tr * sn + ti * cs;
  }
  T2[2 * n] = are; T2[2 * n + 1] = aim;
}

// T2[x][ky][kz][c] -> XF[kx16][ky][kz][c]
__global__ void k_dft_x(const float* __restrict__ T2, float* __restrict__ XF)
{
  int n = blockIdx.x * blockDim.x + threadIdx.x;
  if (n >= 16 * 16 * MMODE * CCH) return;
  int c = n % CCH; int t = n / CCH;
  int kz = t & 7; t >>= 3;
  int ky = t & 15; int kx = t >> 4;
  int fx = (kx < 8) ? kx : (kx + 16);
  float are = 0.f, aim = 0.f;
  for (int x = 0; x < GSZ; ++x) {
    float sn, cs; __sincosf(-(PI_F / 16.0f) * (float)(fx * x), &sn, &cs);
    size_t s = 2 * (((size_t)(x * 16 + ky) * MMODE + kz) * CCH + c);
    float tr = T2[s], ti = T2[s + 1];
    are += tr * cs - ti * sn;
    aim += tr * sn + ti * cs;
  }
  XF[2 * n] = are; XF[2 * n + 1] = aim;
}

// complex per-mode 86x86 matvec; weights differ per mode -> stays VALU
__global__ __launch_bounds__(128)
void k_specmul(const float* __restrict__ XF, const float* __restrict__ wr,
               const float* __restrict__ wi, float* __restrict__ XO)
{
  __shared__ float sre[CCH], sim[CCH];
  int m = blockIdx.x;                 // 0..2047 mode points
  int kz = m & 7, ky = (m >> 3) & 15, kx = m >> 7;
  int corner = ((kx >= 8) ? 2 : 0) + ((ky >= 8) ? 1 : 0);
  int mode = ((kx & 7) * 8 + (ky & 7)) * 8 + kz;
  size_t base = (size_t)m * CCH;
  for (int i = threadIdx.x; i < CCH; i += blockDim.x) {
    sre[i] = XF[2 * (base + i)];
    sim[i] = XF[2 * (base + i) + 1];
  }
  __syncthreads();
  int o = threadIdx.x;
  if (o < CCH) {
    float ar = 0.f, ai = 0.f;
    for (int i = 0; i < CCH; ++i) {
      size_t widx = (((size_t)corner * CCH + i) * CCH + o) * 512 + mode;
      float r = wr[widx], im = wi[widx];
      ar += sre[i] * r - sim[i] * im;
      ai += sre[i] * im + sim[i] * r;
    }
    XO[2 * (base + o)] = ar;
    XO[2 * (base + o) + 1] = ai;
  }
}

// --- truncated inverse DFT, separable ---
__global__ void k_idft_x(const float* __restrict__ XO, float* __restrict__ U1)
{
  int n = blockIdx.x * blockDim.x + threadIdx.x;
  if (n >= GSZ * 16 * MMODE * CCH) return;
  int c = n % CCH; int t = n / CCH;
  int kz = t & 7; t >>= 3;
  int ky = t & 15; int x = t >> 4;
  float are = 0.f, aim = 0.f;
  for (int kx = 0; kx < 16; ++kx) {
    int fx = (kx < 8) ? kx : (kx + 16);
    float sn, cs; __sincosf((PI_F / 16.0f) * (float)(fx * x), &sn, &cs);
    size_t s = 2 * (((size_t)(kx * 16 + ky) * MMODE + kz) * CCH + c);
    float tr = XO[s], ti = XO[s + 1];
    are += tr * cs - ti * sn;
    aim += tr * sn + ti * cs;
  }
  U1[2 * n] = are; U1[2 * n + 1] = aim;
}

__global__ void k_idft_y(const float* __restrict__ U1, float* __restrict__ U2)
{
  int n = blockIdx.x * blockDim.x + threadIdx.x;
  if (n >= GSZ * GSZ * MMODE * CCH) return;
  int c = n % CCH; int t = n / CCH;
  int kz = t & 7; t >>= 3;
  int y = t & 31; int x = t >> 5;
  float are = 0.f, aim = 0.f;
  for (int ky = 0; ky < 16; ++ky) {
    int fy = (ky < 8) ? ky : (ky + 16);
    float sn, cs; __sincosf((PI_F / 16.0f) * (float)(fy * y), &sn, &cs);
    size_t s = 2 * (((size_t)(x * 16 + ky) * MMODE + kz) * CCH + c);
    float tr = U1[s], ti = U1[s + 1];
    are += tr * cs - ti * sn;
    aim += tr * sn + ti * cs;
  }
  U2[2 * n] = are; U2[2 * n + 1] = aim;
}

// irfft along z; writes padded f16 channel stride CP with zeroed pad columns
__global__ void k_idft_z(const float* __restrict__ U2, h16* __restrict__ vspec)
{
  int n = blockIdx.x * blockDim.x + threadIdx.x;
  if (n >= NG * CP) return;
  int c = n % CP; int p = n / CP;
  if (c >= CCH) { vspec[n] = (h16)0.0f; return; }
  int z = p & 31; int y = (p >> 5) & 31; int x = p >> 10;
  size_t rowb = (size_t)(x * GSZ + y) * MMODE;
  float acc = U2[2 * (rowb * CCH + c)];     // kz = 0 (real part)
  for (int kz = 1; kz < MMODE; ++kz) {
    float sn, cs; __sincosf((PI_F / 16.0f) * (float)(kz * z), &sn, &cs);
    size_t s = 2 * ((rowb + kz) * CCH + c);
    acc += 2.0f * (U2[s] * cs - U2[s + 1] * sn);
  }
  vspec[n] = (h16)(acc * (1.0f / 32768.0f));
}

// radius neighbor search on the regular 32^3 lattice (radius 0.055 => +/-2 cells)
__global__ void k_neighbors(const float* __restrict__ xin, int* __restrict__ idx,
                            float* __restrict__ cnt)
{
  int q = blockIdx.x * blockDim.x + threadIdx.x;
  if (q >= NIN) return;
  float qx = xin[q * 3], qy = xin[q * 3 + 1], qz = xin[q * 3 + 2];
  const float R2 = RADF * RADF;
  const float inv_h = 31.0f, h = 1.0f / 31.0f;
  int lox = max(0, (int)ceilf((qx - RADF) * inv_h));
  int hix = min(31, (int)floorf((qx + RADF) * inv_h));
  int loy = max(0, (int)ceilf((qy - RADF) * inv_h));
  int hiy = min(31, (int)floorf((qy + RADF) * inv_h));
  int loz = max(0, (int)ceilf((qz - RADF) * inv_h));
  int hiz = min(31, (int)floorf((qz + RADF) * inv_h));
  int count = 0;
  for (int ix = lox; ix <= hix; ++ix) {
    float dx = qx - (float)ix * h;
    for (int iy = loy; iy <= hiy; ++iy) {
      float dy = qy - (float)iy * h;
      for (int iz = loz; iz <= hiz; ++iz) {
        float dz = qz - (float)iz * h;
        float d2 = dx * dx + dy * dy + dz * dz;
        if (d2 <= R2 && count < KNBR) {
          idx[q * KNBR + count] = (ix * GSZ + iy) * GSZ + iz;
          ++count;
        }
      }
    }
  }
  for (int k = count; k < KNBR; ++k) idx[q * KNBR + k] = -1;
  cnt[q] = (float)max(count, 1);
}

// kernel-MLP input rows: [y_emb[neighbor] (48) | x_emb[query] (48)]
__global__ void k_gather(const h16* __restrict__ yemb, const h16* __restrict__ xemb,
                         const int* __restrict__ idx, h16* __restrict__ kin, int qbase)
{
  int n = blockIdx.x * blockDim.x + threadIdx.x;
  if (n >= ROWSC * 96) return;
  int col = n % 96; int r = n / 96;
  int qloc = r >> 5; int k = r & 31;
  int q = qbase + qloc;
  if (col < 48) {
    int j = idx[q * KNBR + k];
    kin[n] = (j >= 0) ? yemb[(size_t)j * 48 + col] : (h16)0.0f;
  } else {
    kin[n] = xemb[(size_t)q * 48 + (col - 48)];
  }
}

// agg[q][c] = sum_k( h3[row][c] * f_y[idx][c] ) / cnt[q]; padded cols zeroed
__global__ void k_msgmean(const h16* __restrict__ h3, const h16* __restrict__ fy,
                          const int* __restrict__ idx, const float* __restrict__ cnt,
                          h16* __restrict__ agg, int qbase)
{
  int n = blockIdx.x * blockDim.x + threadIdx.x;
  if (n >= CHQ * CP) return;
  int c = n % CP; int qloc = n / CP; int q = qbase + qloc;
  if (c >= CCH) { agg[(size_t)q * CP + c] = (h16)0.0f; return; }
  float s = 0.f;
  for (int k = 0; k < KNBR; ++k) {
    int j = idx[q * KNBR + k];
    if (j >= 0) s += (float)h3[(size_t)(qloc * KNBR + k) * CP + c] * (float)fy[(size_t)j * CP + c];
  }
  agg[(size_t)q * CP + c] = (h16)(s / cnt[q]);
}

// extract column 0 of the padded [NIN,32] projection into d_out (f32)
__global__ void k_out(const h16* __restrict__ pout, float* __restrict__ out)
{
  int q = blockIdx.x * blockDim.x + threadIdx.x;
  if (q < NIN) out[q] = (float)pout[(size_t)q * 32];
}

// =====================================================================
// Host-side orchestration
// =====================================================================
static inline void launch_gemm(const h16* A, const h16* B, const h16* bias,
                               const h16* addend, h16* D, int M, int N, int K,
                               int act, hipStream_t s)
{
  dim3 g((unsigned)(M / BM), (unsigned)(N / BN));
  gemm_f16_wmma<<<g, dim3(256), 0, s>>>(A, B, bias, addend, D, M, N, K, act);
}

extern "C" void kernel_launch(void* const* d_in, const int* in_sizes, int n_in,
                              void* d_out, int out_size, void* d_ws, size_t ws_size,
                              hipStream_t stream)
{
  (void)in_sizes; (void)n_in; (void)out_size; (void)ws_size;
  const float* x_in   = (const float*)d_in[0];
  const float* x_out  = (const float*)d_in[1];
  const float* df     = (const float*)d_in[2];
  const float* W_lift = (const float*)d_in[3];
  const float* b_lift = (const float*)d_in[4];
  const float* sw_r   = (const float*)d_in[5];
  const float* sw_i   = (const float*)d_in[6];
  const float* W_skip = (const float*)d_in[7];
  const float* b_skip = (const float*)d_in[8];
  const float* W_m1   = (const float*)d_in[9];
  const float* b_m1   = (const float*)d_in[10];
  const float* W_m2   = (const float*)d_in[11];
  const float* b_m2   = (const float*)d_in[12];
  const float* W_ms   = (const float*)d_in[13];
  const float* b_ms   = (const float*)d_in[14];
  const float* Wk1    = (const float*)d_in[15];
  const float* bk1    = (const float*)d_in[16];
  const float* Wk2    = (const float*)d_in[17];
  const float* bk2    = (const float*)d_in[18];
  const float* Wk3    = (const float*)d_in[19];
  const float* bk3    = (const float*)d_in[20];
  const float* Wp1    = (const float*)d_in[21];
  const float* bp1    = (const float*)d_in[22];
  const float* Wp2    = (const float*)d_in[23];
  const float* bp2    = (const float*)d_in[24];

  char* base = (char*)d_ws;
  size_t off = 0;
  auto allocB = [&](size_t bytes) { void* p = base + off; off = (off + bytes + 255) & ~(size_t)255; return p; };
  auto allocF = [&](size_t nf) { return (float*)allocB(nf * 4); };
  auto allocH = [&](size_t nh) { return (h16*)allocB(nh * 2); };

  // f16 activations
  h16* y_emb = allocH((size_t)NG * EMB3);
  h16* x_emb = allocH((size_t)NIN * EMB3);
  h16* vin   = allocH((size_t)NG * 32);
  h16* vbuf  = allocH((size_t)NG * CP);
  h16* vspec = allocH((size_t)NG * CP);
  h16* sbuf  = allocH((size_t)NG * CP);
  h16* t1    = allocH((size_t)NG * CP);
  h16* t2    = allocH((size_t)NG * CP);
  h16* kin   = allocH((size_t)ROWSC * 96);
  h16* h1    = allocH((size_t)ROWSC * 512);
  h16* h2    = allocH((size_t)ROWSC * 256);
  h16* h3    = allocH((size_t)ROWSC * CP);
  h16* agg   = allocH((size_t)NIN * CP);
  h16* p1    = allocH((size_t)NIN * 256);
  h16* pout  = allocH((size_t)NIN * 32);
  // f32 spectral intermediates
  float* T1  = allocF((size_t)GSZ * GSZ * MMODE * CCH * 2);
  float* T2  = allocF((size_t)GSZ * 16 * MMODE * CCH * 2);
  float* XF  = allocF((size_t)16 * 16 * MMODE * CCH * 2);
  float* XO  = allocF((size_t)16 * 16 * MMODE * CCH * 2);
  float* U1  = allocF((size_t)GSZ * 16 * MMODE * CCH * 2);
  float* U2  = allocF((size_t)GSZ * GSZ * MMODE * CCH * 2);
  // graph
  int*   idxb = (int*)allocB((size_t)NIN * KNBR * 4);
  float* cntb = allocF((size_t)NIN);
  // padded f16 weights / biases
  h16* pW_lift = allocH(32 * CP);
  h16* pW_skip = allocH((size_t)NL * CP * CP);
  h16* pW_m1   = allocH((size_t)NL * CP * CP);
  h16* pW_m2   = allocH((size_t)NL * CP * CP);
  h16* pW_ms   = allocH((size_t)NL * CP * CP);
  h16* pb_lift = allocH(CP);
  h16* pb_skip = allocH((size_t)NL * CP);
  h16* pb_m1   = allocH((size_t)NL * CP);
  h16* pb_m2   = allocH((size_t)NL * CP);
  h16* pb_ms   = allocH((size_t)NL * CP);
  h16* pWk1    = allocH((size_t)96 * 512);
  h16* pbk1    = allocH(512);
  h16* pWk2    = allocH((size_t)512 * 256);
  h16* pbk2    = allocH(256);
  h16* pWk3    = allocH((size_t)256 * CP);
  h16* pbk3    = allocH(CP);
  h16* pWp1    = allocH((size_t)CP * 256);
  h16* pbp1    = allocH(256);
  h16* pWp2    = allocH((size_t)256 * 32);
  h16* pbp2    = allocH(32);

  const int TB = 256;
  auto nb = [](size_t n) { return (unsigned)((n + 255) / 256); };
  auto pad = [&](const float* s, h16* d, int bn, int r, int c, int pr, int pc) {
    k_padbatch<<<nb((size_t)bn * pr * pc), TB, 0, stream>>>(s, d, bn, r, c, pr, pc);
  };

  // ---- stage padded f16 weights/biases (zero pads self-propagate) ----
  pad(W_lift, pW_lift, 1, 5, CCH, 32, CP);
  pad(W_skip, pW_skip, NL, CCH, CCH, CP, CP);
  pad(W_m1,   pW_m1,   NL, CCH, CCH, CP, CP);
  pad(W_m2,   pW_m2,   NL, CCH, CCH, CP, CP);
  pad(W_ms,   pW_ms,   NL, CCH, CCH, CP, CP);
  pad(b_lift, pb_lift, 1, 1, CCH, 1, CP);
  pad(b_skip, pb_skip, NL, 1, CCH, 1, CP);
  pad(b_m1,   pb_m1,   NL, 1, CCH, 1, CP);
  pad(b_m2,   pb_m2,   NL, 1, CCH, 1, CP);
  pad(b_ms,   pb_ms,   NL, 1, CCH, 1, CP);
  pad(Wk1,    pWk1,    1, 96, 512, 96, 512);
  pad(bk1,    pbk1,    1, 1, 512, 1, 512);
  pad(Wk2,    pWk2,    1, 512, 256, 512, 256);
  pad(bk2,    pbk2,    1, 1, 256, 1, 256);
  pad(Wk3,    pWk3,    1, 256, CCH, 256, CP);
  pad(bk3,    pbk3,    1, 1, CCH, 1, CP);
  pad(Wp1,    pWp1,    1, CCH, 256, CP, 256);
  pad(bp1,    pbp1,    1, 1, 256, 1, 256);
  pad(Wp2,    pWp2,    1, 256, 1, 256, 32);
  pad(bp2,    pbp2,    1, 1, 1, 1, 32);

  // ---- positional embeddings ----
  k_embed<<<nb((size_t)NG * EMB3), TB, 0, stream>>>(x_out, y_emb, NG);
  k_embed<<<nb((size_t)NIN * EMB3), TB, 0, stream>>>(x_in, x_emb, NIN);

  // ---- FNO lifting: [NG,32] x [32,96] ----
  k_build_vin<<<nb((size_t)NG * 32), TB, 0, stream>>>(df, x_out, vin);
  launch_gemm(vin, pW_lift, pb_lift, nullptr, vbuf, NG, CP, 32, 0, stream);

  // ---- 4 spectral blocks ----
  for (int l = 0; l < NL; ++l) {
    const size_t wl = (size_t)l * 4 * CCH * CCH * 512;
    k_dft_z<<<nb((size_t)GSZ * GSZ * MMODE * CCH), TB, 0, stream>>>(vbuf, T1);
    k_dft_y<<<nb((size_t)GSZ * 16 * MMODE * CCH), TB, 0, stream>>>(T1, T2);
    k_dft_x<<<nb((size_t)16 * 16 * MMODE * CCH), TB, 0, stream>>>(T2, XF);
    k_specmul<<<16 * 16 * MMODE, 128, 0, stream>>>(XF, sw_r + wl, sw_i + wl, XO);
    k_idft_x<<<nb((size_t)GSZ * 16 * MMODE * CCH), TB, 0, stream>>>(XO, U1);
    k_idft_y<<<nb((size_t)GSZ * GSZ * MMODE * CCH), TB, 0, stream>>>(U1, U2);
    k_idft_z<<<nb((size_t)NG * CP), TB, 0, stream>>>(U2, vspec);
    // s = gelu(spectral + skip)
    launch_gemm(vbuf, pW_skip + (size_t)l * CP * CP, pb_skip + l * CP, vspec,
                sbuf, NG, CP, CP, 1, stream);
    // channel MLP + shortcut
    launch_gemm(sbuf, pW_m1 + (size_t)l * CP * CP, pb_m1 + l * CP, nullptr,
                t1, NG, CP, CP, 1, stream);
    launch_gemm(sbuf, pW_ms + (size_t)l * CP * CP, pb_ms + l * CP, nullptr,
                t2, NG, CP, CP, 0, stream);
    launch_gemm(t1, pW_m2 + (size_t)l * CP * CP, pb_m2 + l * CP, t2,
                vbuf, NG, CP, CP, (l < NL - 1) ? 1 : 0, stream);
  }
  // f_y == vbuf, [NG, CP] f16 row-major in (x,y,z) order

  // ---- GNO: lattice radius search + chunked kernel-MLP (WMMA GEMMs) ----
  k_neighbors<<<nb(NIN), TB, 0, stream>>>(x_in, idxb, cntb);
  for (int ch = 0; ch < NCHUNK; ++ch) {
    int qbase = ch * CHQ;
    k_gather<<<nb((size_t)ROWSC * 96), TB, 0, stream>>>(y_emb, x_emb, idxb, kin, qbase);
    launch_gemm(kin, pWk1, pbk1, nullptr, h1, ROWSC, 512, 96, 1, stream);
    launch_gemm(h1, pWk2, pbk2, nullptr, h2, ROWSC, 256, 512, 1, stream);
    launch_gemm(h2, pWk3, pbk3, nullptr, h3, ROWSC, CP, 256, 0, stream);
    k_msgmean<<<nb((size_t)CHQ * CP), TB, 0, stream>>>(h3, vbuf, idxb, cntb, agg, qbase);
  }

  // ---- projection head ----
  launch_gemm(agg, pWp1, pbp1, nullptr, p1, NIN, 256, CP, 1, stream);
  launch_gemm(p1, pWp2, pbp2, nullptr, pout, NIN, 32, 256, 0, stream);
  k_out<<<nb(NIN), TB, 0, stream>>>(pout, (float*)d_out);
}
